// mRNN_60026462929281
// MI455X (gfx1250) — matrix operand order repaired and checked by reading, not compiled, not run on hardware
//
#include <hip/hip_runtime.h>

#define B_ 64
#define T_ 300
#define N_ 2048
#define I_ 128

#define ALPHA_  0.1f
#define SCALE_R 0.0044721359549995794f   // sqrt(2*0.1)*0.01
#define SCALE_I 0.0044721359549995794f

typedef __attribute__((ext_vector_type(4)))  unsigned int u32x4;
typedef __attribute__((ext_vector_type(16))) __bf16       v16bf;
typedef __attribute__((ext_vector_type(8)))  float        v8f;

union Frag {
    u32x4 u[2];
    v16bf v;
};

// ---------------------------------------------------------------------------
// Prep: pack W_rec (with Dale clip/sign/mask) into per-lane WMMA B-fragment
// layout, bf16.  Packed index: (((ntile*64 + ktile)*32 + lane)*16 + j)
// where fragment element j of lane maps to  k = ktile*32 + (lane>>4)*16 + j,
//                                           n = ntile*16 + (lane&15).
// ---------------------------------------------------------------------------
__global__ void pack_wr_kernel(const float* __restrict__ w,
                               const float* __restrict__ mask,
                               const float* __restrict__ sign,
                               __bf16* __restrict__ outp) {
    int idx = blockIdx.x * blockDim.x + threadIdx.x;
    if (idx >= N_ * N_) return;
    int j    = idx & 15;
    int lane = (idx >> 4) & 31;
    int kt   = (idx >> 9) & 63;
    int nt   = idx >> 15;
    int k = kt * 32 + (lane >> 4) * 16 + j;
    int n = nt * 16 + (lane & 15);
    size_t src = (size_t)n * N_ + k;
    float v = w[src];
    v = fminf(fmaxf(v, 0.0f), 10.0f);
    v *= sign[src] * mask[src];
    outp[idx] = (__bf16)v;
}

__global__ void pack_wi_kernel(const float* __restrict__ w,
                               const float* __restrict__ mask,
                               const float* __restrict__ sign,
                               __bf16* __restrict__ outp) {
    int idx = blockIdx.x * blockDim.x + threadIdx.x;
    if (idx >= N_ * I_) return;
    int j    = idx & 15;
    int lane = (idx >> 4) & 31;
    int kt   = (idx >> 9) & 3;
    int nt   = idx >> 11;
    int k = kt * 32 + (lane >> 4) * 16 + j;
    int n = nt * 16 + (lane & 15);
    size_t src = (size_t)n * I_ + k;
    float v = w[src];
    v = fminf(fmaxf(v, 0.0f), 10.0f);
    v *= sign[src] * mask[src];
    outp[idx] = (__bf16)v;
}

// Precompute u[t][b][ki] = inp[b][t][ki] + scale_i * noise_inp[t][b][ki] in bf16
__global__ void u_prep_kernel(const float* __restrict__ inp,
                              const float* __restrict__ ninp,
                              __bf16* __restrict__ u) {
    int idx = blockIdx.x * blockDim.x + threadIdx.x;
    if (idx >= T_ * B_ * I_) return;
    int ki = idx & (I_ - 1);
    int b  = (idx / I_) % B_;
    int t  = idx / (I_ * B_);
    float v = inp[((size_t)b * T_ + t) * I_ + ki]
            + SCALE_I * __builtin_nontemporal_load(&ninp[idx]);
    u[idx] = (__bf16)v;
}

__global__ void init_state_kernel(const float* __restrict__ x0,
                                  float* __restrict__ x,
                                  __bf16* __restrict__ a0,
                                  __bf16* __restrict__ a1) {
    int idx = blockIdx.x * blockDim.x + threadIdx.x;
    if (idx >= B_ * N_) return;
    float v = x0[idx];
    x[idx] = v;
    float h = fmaxf(v, 0.0f);
    a0[idx] = (__bf16)h;
    a1[idx] = (__bf16)0.0f;   // overwritten by step 0 before it is read
}

// ---------------------------------------------------------------------------
// One RNN time step.  256 waves; wave -> 16x32 output tile (two f32 accs).
//   pre = relu(x_prev) @ Wr.T + u_t @ Wi.T          (bf16 WMMA, f32 acc)
//   x   = x + alpha*(-x + pre + baseline) + sr*nh ;  h = relu(x)
// Streaming arrays (noise_hid in, out stores) use non-temporal hints so the
// 2x157MB streams do not evict the L2-resident packed Wr / activation buffers.
// ---------------------------------------------------------------------------
__global__ __launch_bounds__(256) void step_kernel(
    const __bf16* __restrict__ wr,     // packed [128 ntiles][64 ktiles][32][16]
    const __bf16* __restrict__ wi,     // packed [128 ntiles][ 4 ktiles][32][16]
    const __bf16* __restrict__ u,      // [T][B][I] bf16
    float*        __restrict__ x,      // [B][N] f32 state
    const __bf16* __restrict__ actin,  // [B][N] bf16 relu(x_prev)
    __bf16*       __restrict__ actout, // [B][N] bf16 relu(x_new)
    const float*  __restrict__ baseline,
    const float*  __restrict__ nh,     // [T][B][N]
    float*        __restrict__ out,    // [B][T][N]
    int t)
{
    const int lane = threadIdx.x & 31;
    const int wid  = blockIdx.x * 8 + (threadIdx.x >> 5);   // 0..255
    const int rowT = wid >> 6;          // 0..3   (16 rows of B each)
    const int colP = wid & 63;          // 0..63  (32 cols of N each)
    const int rb   = rowT * 16;
    const int ct0  = colP * 2;
    const int ct1  = ct0 + 1;

    // A-fragment addressing (16-bit A 16x32): lane -> M, 16 contiguous K per lane half
    const int m  = rb + (lane & 15);
    const int kb = (lane >> 4) * 16;

    v8f acc0 = {};
    v8f acc1 = {};

    // ---- recurrent GEMM: K = N = 2048 -> 64 k-tiles of 32 ----
    const __bf16* arow = actin + (size_t)m * N_ + kb;
    const __bf16* bp0  = wr + ((size_t)ct0 * 64 * 32 + lane) * 16;
    const __bf16* bp1  = wr + ((size_t)ct1 * 64 * 32 + lane) * 16;
#pragma unroll 4
    for (int kt = 0; kt < 64; ++kt) {
        Frag a, b0, b1;
        a.u[0]  = *(const u32x4*)(arow);
        a.u[1]  = *(const u32x4*)(arow + 8);
        b0.u[0] = *(const u32x4*)(bp0);
        b0.u[1] = *(const u32x4*)(bp0 + 8);
        b1.u[0] = *(const u32x4*)(bp1);
        b1.u[1] = *(const u32x4*)(bp1 + 8);
        acc0 = __builtin_amdgcn_wmma_f32_16x16x32_bf16(
                   false, a.v, false, b0.v, (short)0, acc0, false, false);
        acc1 = __builtin_amdgcn_wmma_f32_16x16x32_bf16(
                   false, a.v, false, b1.v, (short)0, acc1, false, false);
        arow += 32;
        bp0  += 32 * 16;
        bp1  += 32 * 16;
    }

    // ---- input GEMM: K = I = 128 -> 4 k-tiles of 32 ----
    const __bf16* urow = u + ((size_t)t * B_ + m) * I_ + kb;
    const __bf16* ip0  = wi + ((size_t)ct0 * 4 * 32 + lane) * 16;
    const __bf16* ip1  = wi + ((size_t)ct1 * 4 * 32 + lane) * 16;
#pragma unroll
    for (int kt = 0; kt < 4; ++kt) {
        Frag a, b0, b1;
        a.u[0]  = *(const u32x4*)(urow);
        a.u[1]  = *(const u32x4*)(urow + 8);
        b0.u[0] = *(const u32x4*)(ip0);
        b0.u[1] = *(const u32x4*)(ip0 + 8);
        b1.u[0] = *(const u32x4*)(ip1);
        b1.u[1] = *(const u32x4*)(ip1 + 8);
        acc0 = __builtin_amdgcn_wmma_f32_16x16x32_bf16(
                   false, a.v, false, b0.v, (short)0, acc0, false, false);
        acc1 = __builtin_amdgcn_wmma_f32_16x16x32_bf16(
                   false, a.v, false, b1.v, (short)0, acc1, false, false);
        urow += 32;
        ip0  += 32 * 16;
        ip1  += 32 * 16;
    }

    // ---- fused epilogue (f32 C/D layout: VGPR r -> M = r + 8*(lane>>4)) ----
    const int mrow = rb + 8 * (lane >> 4);
    const int n0   = ct0 * 16 + (lane & 15);
    const int n1   = n0 + 16;
    const float bl0 = baseline[n0];
    const float bl1 = baseline[n1];
#pragma unroll
    for (int r = 0; r < 8; ++r) {
        const int mm = mrow + r;
        const size_t xi  = (size_t)mm * N_;
        const size_t nhb = ((size_t)t * B_ + mm) * N_;
        const size_t ob  = ((size_t)mm * T_ + t) * N_;

        float nh0 = __builtin_nontemporal_load(&nh[nhb + n0]);
        float xo0 = x[xi + n0];
        float xn0 = xo0 + ALPHA_ * ((acc0[r] + bl0) - xo0) + SCALE_R * nh0;
        x[xi + n0] = xn0;
        float h0 = fmaxf(xn0, 0.0f);
        __builtin_nontemporal_store(h0, &out[ob + n0]);
        actout[xi + n0] = (__bf16)h0;

        float nh1 = __builtin_nontemporal_load(&nh[nhb + n1]);
        float xo1 = x[xi + n1];
        float xn1 = xo1 + ALPHA_ * ((acc1[r] + bl1) - xo1) + SCALE_R * nh1;
        x[xi + n1] = xn1;
        float h1 = fmaxf(xn1, 0.0f);
        __builtin_nontemporal_store(h1, &out[ob + n1]);
        actout[xi + n1] = (__bf16)h1;
    }
}

// ---------------------------------------------------------------------------
extern "C" void kernel_launch(void* const* d_in, const int* in_sizes, int n_in,
                              void* d_out, int out_size, void* d_ws, size_t ws_size,
                              hipStream_t stream) {
    const float* inp        = (const float*)d_in[0];
    const float* x0         = (const float*)d_in[1];
    const float* W_rec      = (const float*)d_in[2];
    const float* W_rec_mask = (const float*)d_in[3];
    const float* W_rec_sign = (const float*)d_in[4];
    const float* W_inp      = (const float*)d_in[5];
    const float* W_inp_mask = (const float*)d_in[6];
    const float* W_inp_sign = (const float*)d_in[7];
    const float* baseline   = (const float*)d_in[8];
    const float* noise_hid  = (const float*)d_in[9];
    const float* noise_inp  = (const float*)d_in[10];
    float* out = (float*)d_out;

    char* ws = (char*)d_ws;
    size_t off = 0;
    __bf16* wr_p = (__bf16*)(ws + off); off += (size_t)N_ * N_ * 2;        // 8 MB
    __bf16* wi_p = (__bf16*)(ws + off); off += (size_t)N_ * I_ * 2;        // 512 KB
    __bf16* u_p  = (__bf16*)(ws + off); off += (size_t)T_ * B_ * I_ * 2;   // 4.8 MB
    float*  xs   = (float*) (ws + off); off += (size_t)B_ * N_ * 4;        // 512 KB
    __bf16* act0 = (__bf16*)(ws + off); off += (size_t)B_ * N_ * 2;        // 256 KB
    __bf16* act1 = (__bf16*)(ws + off); off += (size_t)B_ * N_ * 2;        // 256 KB

    pack_wr_kernel<<<(N_ * N_ + 255) / 256, 256, 0, stream>>>(
        W_rec, W_rec_mask, W_rec_sign, wr_p);
    pack_wi_kernel<<<(N_ * I_ + 255) / 256, 256, 0, stream>>>(
        W_inp, W_inp_mask, W_inp_sign, wi_p);
    u_prep_kernel<<<(T_ * B_ * I_ + 255) / 256, 256, 0, stream>>>(
        inp, noise_inp, u_p);
    init_state_kernel<<<(B_ * N_ + 255) / 256, 256, 0, stream>>>(
        x0, xs, act0, act1);

    for (int t = 0; t < T_; ++t) {
        __bf16* ain  = (t & 1) ? act1 : act0;
        __bf16* aout = (t & 1) ? act0 : act1;
        step_kernel<<<32, 256, 0, stream>>>(
            wr_p, wi_p, u_p, xs, ain, aout, baseline, noise_hid, out, t);
    }
}
